// SpatialGATLargeWithJointEncoder_30099130810806
// MI455X (gfx1250) — matrix operand order repaired and checked by reading, not compiled, not run on hardware
//
#include <hip/hip_runtime.h>
#include <hip/hip_bf16.h>
#include <cstdint>

typedef __attribute__((ext_vector_type(16))) _Float16 v16h;
typedef __attribute__((ext_vector_type(8)))  _Float16 v8h;
typedef __attribute__((ext_vector_type(8)))  float    v8f;

#define LN_EPS 1e-5f

// ---------------------------------------------------------------------------
// GEMM: C[MxN] = act(A[MxK] @ B[KxN] + bias), all fp32 in memory.
// Split-fp16 WMMA (Markidis): x = hi + lo (fp16 each);
//   A*B ~= Ahi*Bhi + Alo*Bhi + Ahi*Blo, fp32 accumulate -> ~fp32 accuracy.
// Block tile 128x128, 256 threads (8 waves, 2x4), wave tile 64x32.
// ---------------------------------------------------------------------------
template<int ACT>  // 0 = none, 1 = relu
__global__ __launch_bounds__(256) void k_gemm(
    const float* __restrict__ A, const float* __restrict__ B,
    const float* __restrict__ bias, float* __restrict__ C,
    int M, int N, int K)
{
  __shared__ alignas(32) _Float16 Ah[128][32];
  __shared__ alignas(32) _Float16 Al[128][32];
  __shared__ alignas(32) _Float16 Bh[128][32];  // transposed: [n][k]
  __shared__ alignas(32) _Float16 Bl[128][32];

  const int tid  = threadIdx.x;
  const int lane = tid & 31;
  const int w    = tid >> 5;
  const int wm   = w >> 2;       // 0..1  (M direction, 64 rows each)
  const int wn   = w & 3;        // 0..3  (N direction, 32 cols each)
  const int hi   = lane >> 4;    // half-wave selector
  const int lr   = lane & 15;
  const int bm   = blockIdx.y * 128;
  const int bn   = blockIdx.x * 128;

  const int lrow = tid >> 1;          // 0..127 staging row
  const int kb   = (tid & 1) * 16;    // k sub-chunk 0 or 16

  v8f acc[4][2];
  const v8f vzero = {0.f,0.f,0.f,0.f,0.f,0.f,0.f,0.f};
#pragma unroll
  for (int i = 0; i < 4; ++i)
#pragma unroll
    for (int j = 0; j < 2; ++j) acc[i][j] = vzero;

  for (int k0 = 0; k0 < K; k0 += 32) {
    // speculative prefetch of next K-tile (global_prefetch_b8)
    __builtin_prefetch(&A[(size_t)(bm + lrow) * K + (k0 + 32 + kb)], 0, 0);

    // ---- stage A (row-major [m][k]) and B (transposed [n][k]) as hi/lo fp16
    {
      const int grow = bm + lrow;
#pragma unroll
      for (int i = 0; i < 16; ++i) {
        const int k = k0 + kb + i;
        float v = (grow < M && k < K) ? A[(size_t)grow * K + k] : 0.0f;
        _Float16 h16 = (_Float16)v;
        Ah[lrow][kb + i] = h16;
        Al[lrow][kb + i] = (_Float16)(v - (float)h16);
      }
      const int gcol = bn + lrow;
#pragma unroll
      for (int i = 0; i < 16; ++i) {
        const int k = k0 + kb + i;
        float v = (gcol < N && k < K) ? B[(size_t)k * N + gcol] : 0.0f;
        _Float16 h16 = (_Float16)v;
        Bh[lrow][kb + i] = h16;
        Bl[lrow][kb + i] = (_Float16)(v - (float)h16);
      }
    }
    __syncthreads();

    // ---- build fragments in contiguous k order (k = 16*hi + e) for both A and B
    v16h ah[4], al[4], bh[2], bl[2];
#pragma unroll
    for (int i = 0; i < 4; ++i) {
      const int r = wm * 64 + i * 16 + lr;       // M row = lane&15
      v8h h0 = *(const v8h*)&Ah[r][hi * 16];      // K = (hi?16:0)+0..7
      v8h h1 = *(const v8h*)&Ah[r][hi * 16 + 8];  // K = (hi?16:0)+8..15
      v8h l0 = *(const v8h*)&Al[r][hi * 16];
      v8h l1 = *(const v8h*)&Al[r][hi * 16 + 8];
#pragma unroll
      for (int e = 0; e < 8; ++e) {
        ah[i][e] = h0[e]; ah[i][8 + e] = h1[e];
        al[i][e] = l0[e]; al[i][8 + e] = l1[e];
      }
    }
#pragma unroll
    for (int j = 0; j < 2; ++j) {
      const int c = wn * 32 + j * 16 + lr;        // N col = lane&15
      bh[j] = *(const v16h*)&Bh[c][hi * 16];      // K = (hi?16:0)+0..15
      bl[j] = *(const v16h*)&Bl[c][hi * 16];
    }

    // ---- 24 WMMAs per K-step: hi*lo + lo*hi + hi*hi correction order
#pragma unroll
    for (int i = 0; i < 4; ++i) {
#pragma unroll
      for (int j = 0; j < 2; ++j) {
        acc[i][j] = __builtin_amdgcn_wmma_f32_16x16x32_f16(
            false, ah[i], false, bl[j], (short)0, acc[i][j], false, false);
        acc[i][j] = __builtin_amdgcn_wmma_f32_16x16x32_f16(
            false, al[i], false, bh[j], (short)0, acc[i][j], false, false);
        acc[i][j] = __builtin_amdgcn_wmma_f32_16x16x32_f16(
            false, ah[i], false, bh[j], (short)0, acc[i][j], false, false);
      }
    }
    __syncthreads();
  }

  // ---- store: D layout => VGPR e -> M = (hi?8:0)+e, N = lane&15
#pragma unroll
  for (int i = 0; i < 4; ++i) {
#pragma unroll
    for (int j = 0; j < 2; ++j) {
      const int col = bn + wn * 32 + j * 16 + lr;
#pragma unroll
      for (int e = 0; e < 8; ++e) {
        const int row = bm + wm * 64 + i * 16 + hi * 8 + e;
        if (row < M && col < N) {
          float v = acc[i][j][e];
          if (bias) v += bias[col];
          if (ACT == 1) v = fmaxf(v, 0.0f);
          C[(size_t)row * N + col] = v;
        }
      }
    }
  }
}

// ---------------------------------------------------------------------------
// LayerNorm(256) + ReLU, in place. One wave per row (wave32: 8 f32/lane).
// ---------------------------------------------------------------------------
__global__ __launch_bounds__(256) void k_ln_relu(
    float* __restrict__ h, const float* __restrict__ g,
    const float* __restrict__ b, int N)
{
  const int wid = threadIdx.x >> 5, lane = threadIdx.x & 31;
  const int row = blockIdx.x * 8 + wid;
  if (row >= N) return;
  float* hp = h + (size_t)row * 256;
  float v[8];
  float s = 0.f;
#pragma unroll
  for (int i = 0; i < 8; ++i) { v[i] = hp[i * 32 + lane]; s += v[i]; }
#pragma unroll
  for (int o = 16; o > 0; o >>= 1) s += __shfl_xor(s, o, 32);
  const float mean = s * (1.0f / 256.0f);
  float q = 0.f;
#pragma unroll
  for (int i = 0; i < 8; ++i) { float d = v[i] - mean; q += d * d; }
#pragma unroll
  for (int o = 16; o > 0; o >>= 1) q += __shfl_xor(q, o, 32);
  const float inv = rsqrtf(q * (1.0f / 256.0f) + LN_EPS);
#pragma unroll
  for (int i = 0; i < 8; ++i) {
    const int c = i * 32 + lane;
    const float o = (v[i] - mean) * inv * g[c] + b[c];
    hp[c] = fmaxf(o, 0.f);
  }
}

// ---------------------------------------------------------------------------
// Attention score projections: a_src/a_dst[n,h] = sum_c h[n,h,c]*att[h,c]
// ---------------------------------------------------------------------------
__global__ __launch_bounds__(256) void k_att(
    const float* __restrict__ h, const float* __restrict__ wsrc,
    const float* __restrict__ wdst, float* __restrict__ asrc,
    float* __restrict__ adst, int N, int H, int Cc)
{
  const int t = blockIdx.x * blockDim.x + threadIdx.x;
  if (t >= N * H) return;
  const int n = t / H, hd = t - n * H;
  const float* hp = h + ((size_t)n * H + hd) * Cc;
  const float* sp = wsrc + (size_t)hd * Cc;
  const float* dp = wdst + (size_t)hd * Cc;
  float s1 = 0.f, s2 = 0.f;
  for (int c = 0; c < Cc; ++c) { const float x = hp[c]; s1 += x * sp[c]; s2 += x * dp[c]; }
  asrc[t] = s1; adst[t] = s2;
}

__global__ void k_fill_u32(unsigned* __restrict__ p, unsigned v, long long n) {
  const long long t = (long long)blockIdx.x * blockDim.x + threadIdx.x;
  if (t < n) p[t] = v;
}

// monotonic float<->uint encoding so atomicMax(u32) == float max (handles negatives)
__device__ __forceinline__ unsigned f2ord(float f) {
  unsigned u = __float_as_uint(f);
  return (u & 0x80000000u) ? ~u : (u | 0x80000000u);
}
__device__ __forceinline__ float ord2f(unsigned u) {
  return (u & 0x80000000u) ? __uint_as_float(u & 0x7FFFFFFFu) : __uint_as_float(~u);
}

__device__ __forceinline__ void edge_sd(const int* ei, long long e, int E, int N,
                                        int& s, int& d) {
  if (e < E) { s = ei[e]; d = ei[(size_t)E + e]; }
  else       { s = (int)(e - E); d = s; }            // self loops
  s = min(max(s, 0), N - 1); d = min(max(d, 0), N - 1);
}

// pass 1: segment max of leaky_relu(a_src[src]+a_dst[dst])
__global__ void k_edge_max(const int* __restrict__ ei, int E, int N, int H,
    const float* __restrict__ asrc, const float* __restrict__ adst,
    unsigned* __restrict__ emax)
{
  const long long t = (long long)blockIdx.x * blockDim.x + threadIdx.x;
  const long long ET = (long long)E + N;
  if (t >= ET * H) return;
  const int hd = (int)(t % H); const long long e = t / H;
  int s, d; edge_sd(ei, e, E, N, s, d);
  float x = asrc[(size_t)s * H + hd] + adst[(size_t)d * H + hd];
  x = (x > 0.f) ? x : 0.2f * x;
  atomicMax(&emax[(size_t)d * H + hd], f2ord(x));
}

// pass 2: ee = exp(e - emax[dst]); denom[dst] += ee
__global__ void k_edge_exp(const int* __restrict__ ei, int E, int N, int H,
    const float* __restrict__ asrc, const float* __restrict__ adst,
    const unsigned* __restrict__ emax, float* __restrict__ denom,
    float* __restrict__ ee)
{
  const long long t = (long long)blockIdx.x * blockDim.x + threadIdx.x;
  const long long ET = (long long)E + N;
  if (t >= ET * H) return;
  const int hd = (int)(t % H); const long long e = t / H;
  int s, d; edge_sd(ei, e, E, N, s, d);
  float x = asrc[(size_t)s * H + hd] + adst[(size_t)d * H + hd];
  x = (x > 0.f) ? x : 0.2f * x;
  const float m = ord2f(emax[(size_t)d * H + hd]);
  const float w = expf(x - m);
  ee[t] = w;
  atomicAdd(&denom[(size_t)d * H + hd], w);
}

// pass 3: out[dst,h,:] += (ee/denom[dst,h]) * feat[src,h,:]  (float4 chunks)
__global__ void k_edge_accum(const int* __restrict__ ei, int E, int N, int H, int Cc,
    const float* __restrict__ feat, const float* __restrict__ ee,
    const float* __restrict__ denom, float* __restrict__ out)
{
  const long long t = (long long)blockIdx.x * blockDim.x + threadIdx.x;
  const int Cv = Cc >> 2;
  const long long ET = (long long)E + N;
  if (t >= ET * H * Cv) return;
  const int cv = (int)(t % Cv);
  const long long r = t / Cv;
  const int hd = (int)(r % H);
  const long long e = r / H;
  int s, d; edge_sd(ei, e, E, N, s, d);
  const float alpha = ee[(size_t)e * H + hd] / (denom[(size_t)d * H + hd] + 1e-16f);
  const float4 hv = *(const float4*)(feat + ((size_t)s * H + hd) * Cc + (size_t)cv * 4);
  float* op = out + ((size_t)d * H + hd) * Cc + (size_t)cv * 4;
  atomicAdd(op + 0, alpha * hv.x);
  atomicAdd(op + 1, alpha * hv.y);
  atomicAdd(op + 2, alpha * hv.z);
  atomicAdd(op + 3, alpha * hv.w);
}

// out = elu(g + bias[col]) + res
__global__ void k_elu_res(const float* __restrict__ g, const float* __restrict__ bias,
    const float* __restrict__ res, float* __restrict__ out, long long total, int W)
{
  const long long t = (long long)blockIdx.x * blockDim.x + threadIdx.x;
  if (t >= total) return;
  const int col = (int)(t % W);
  float x = g[t] + bias[col];
  x = (x > 0.f) ? x : expm1f(x);
  out[t] = x + res[t];
}

// ---------------------------------------------------------------------------
extern "C" void kernel_launch(void* const* d_in, const int* in_sizes, int n_in,
                              void* d_out, int out_size, void* d_ws, size_t ws_size,
                              hipStream_t stream) {
  const float* x        = (const float*)d_in[0];
  const int*   ei       = (const int*)d_in[1];
  const float* enc_w1   = (const float*)d_in[2];
  const float* enc_b1   = (const float*)d_in[3];
  const float* ln1_g    = (const float*)d_in[4];
  const float* ln1_b    = (const float*)d_in[5];
  const float* enc_w2   = (const float*)d_in[6];
  const float* enc_b2   = (const float*)d_in[7];
  const float* ln2_g    = (const float*)d_in[8];
  const float* ln2_b    = (const float*)d_in[9];
  const float* w1       = (const float*)d_in[10];
  const float* att_src1 = (const float*)d_in[11];
  const float* att_dst1 = (const float*)d_in[12];
  const float* bias1    = (const float*)d_in[13];
  const float* res1_w   = (const float*)d_in[14];
  const float* res1_b   = (const float*)d_in[15];
  const float* w2       = (const float*)d_in[16];
  const float* att_src2 = (const float*)d_in[17];
  const float* att_dst2 = (const float*)d_in[18];
  const float* bias2    = (const float*)d_in[19];
  const float* res2_w   = (const float*)d_in[20];
  const float* res2_b   = (const float*)d_in[21];
  const float* w3       = (const float*)d_in[22];
  const float* att_src3 = (const float*)d_in[23];
  const float* att_dst3 = (const float*)d_in[24];
  const float* bias3    = (const float*)d_in[25];
  const float* res3_w   = (const float*)d_in[26];
  const float* res3_b   = (const float*)d_in[27];
  const float* mlp_w1   = (const float*)d_in[28];
  const float* mlp_b1   = (const float*)d_in[29];
  const float* mlp_w2   = (const float*)d_in[30];
  const float* mlp_b2   = (const float*)d_in[31];

  const int G = in_sizes[2] / 256;        // 405
  const int N = in_sizes[0] / G;          // 50000
  const int E = in_sizes[1] / 2;          // 300000
  const long long ET = (long long)E + N;  // edges + self loops

  // ---- workspace carve (256B aligned regions) ----
  char* wp = (char*)d_ws;
  auto carve = [&](size_t bytes) {
    char* p = wp; wp += (bytes + 255) & ~(size_t)255; return (void*)p;
  };
  float*    h0    = (float*)carve((size_t)N * 256 * 4);   // encoder output
  float*    bufA  = (float*)carve((size_t)N * 1024 * 4);  // ping
  float*    bufB  = (float*)carve((size_t)N * 1024 * 4);  // pong (gat accum)
  float*    bufC  = (float*)carve((size_t)N * 1024 * 4);  // residual / features
  float*    asrc  = (float*)carve((size_t)N * 4 * 4);
  float*    adst  = (float*)carve((size_t)N * 4 * 4);
  unsigned* emax  = (unsigned*)carve((size_t)N * 4 * 4);
  float*    denom = (float*)carve((size_t)N * 4 * 4);
  float*    ee    = (float*)carve((size_t)ET * 4 * 4);
  (void)ws_size; (void)n_in; (void)out_size;

  auto gemm = [&](const float* Ap, const float* Bp, const float* bp, float* Cp,
                  int M, int Nn, int K, int act) {
    dim3 grid((Nn + 127) / 128, (M + 127) / 128);
    if (act) k_gemm<1><<<grid, 256, 0, stream>>>(Ap, Bp, bp, Cp, M, Nn, K);
    else     k_gemm<0><<<grid, 256, 0, stream>>>(Ap, Bp, bp, Cp, M, Nn, K);
  };
  auto fill = [&](void* p, unsigned v, long long n) {
    k_fill_u32<<<(unsigned)((n + 255) / 256), 256, 0, stream>>>((unsigned*)p, v, n);
  };
  auto gat = [&](const float* feat, const float* aw_s, const float* aw_d,
                 float* gout, int H, int Cc) {
    k_att<<<(N * H + 255) / 256, 256, 0, stream>>>(feat, aw_s, aw_d, asrc, adst, N, H, Cc);
    fill(emax, 0x007FFFFFu, (long long)N * H);     // f2ord(-inf)
    fill(denom, 0u, (long long)N * H);
    fill(gout, 0u, (long long)N * H * Cc);
    const long long eh = ET * H;
    k_edge_max<<<(unsigned)((eh + 255) / 256), 256, 0, stream>>>(ei, E, N, H, asrc, adst, emax);
    k_edge_exp<<<(unsigned)((eh + 255) / 256), 256, 0, stream>>>(ei, E, N, H, asrc, adst, emax, denom, ee);
    const long long tot = ET * H * (Cc >> 2);
    k_edge_accum<<<(unsigned)((tot + 255) / 256), 256, 0, stream>>>(ei, E, N, H, Cc, feat, ee, denom, gout);
  };

  // ---- JointEncoder ----
  gemm(x, enc_w1, enc_b1, bufA, N, 256, G, 0);
  k_ln_relu<<<(N + 7) / 8, 256, 0, stream>>>(bufA, ln1_g, ln1_b, N);
  gemm(bufA, enc_w2, enc_b2, h0, N, 256, 256, 0);
  k_ln_relu<<<(N + 7) / 8, 256, 0, stream>>>(h0, ln2_g, ln2_b, N);

  // ---- Block 1: 256 -> 4 heads x 256 (concat) ----
  gemm(h0, w1, nullptr, bufA, N, 1024, 256, 0);                  // features h = idn @ W
  gat(bufA, att_src1, att_dst1, bufB, 4, 256);                   // bufB = segment_sum
  gemm(h0, res1_w, res1_b, bufC, N, 1024, 256, 0);               // residual
  k_elu_res<<<(unsigned)(((long long)N * 1024 + 255) / 256), 256, 0, stream>>>(
      bufB, bias1, bufC, bufA, (long long)N * 1024, 1024);       // bufA = h1

  // ---- Block 2: 1024 -> 2 heads x 96 (concat) ----
  gemm(bufA, w2, nullptr, bufC, N, 192, 1024, 0);
  gat(bufC, att_src2, att_dst2, bufB, 2, 96);
  gemm(bufA, res2_w, res2_b, bufC, N, 192, 1024, 0);
  k_elu_res<<<(unsigned)(((long long)N * 192 + 255) / 256), 256, 0, stream>>>(
      bufB, bias2, bufC, bufA, (long long)N * 192, 192);         // bufA = h2

  // ---- Block 3: 192 -> 128 (1 head, mean == identity) ----
  gemm(bufA, w3, nullptr, bufC, N, 128, 192, 0);
  gat(bufC, att_src3, att_dst3, bufB, 1, 128);
  gemm(bufA, res3_w, res3_b, bufC, N, 128, 192, 0);
  k_elu_res<<<(unsigned)(((long long)N * 128 + 255) / 256), 256, 0, stream>>>(
      bufB, bias3, bufC, bufA, (long long)N * 128, 128);         // bufA = h3

  // ---- MLP head ----
  gemm(bufA, mlp_w1, mlp_b1, bufB, N, 64, 128, 1);               // relu
  gemm(bufB, mlp_w2, mlp_b2, (float*)d_out, N, 2, 64, 0);
}
